// MultiHeadSelfAttention_18141941858661
// MI455X (gfx1250) — matrix operand order repaired and checked by reading, not compiled, not run on hardware
//
#include <hip/hip_runtime.h>

// MHA forward for B=4, S=2048, D=768, H=12, DK=64 on gfx1250 (wave32, WMMA, TDM).
#define BATCH  4
#define SEQ    2048
#define DMODEL 768
#define NHEAD  12
#define DHEAD  64
#define MROWS  (BATCH * SEQ)   // 8192
#define KCHUNK 32              // keys per staged chunk
#define NCHUNK (SEQ / KCHUNK)  // 64

typedef __attribute__((ext_vector_type(16))) _Float16 v16h;
typedef __attribute__((ext_vector_type(8)))  _Float16 v8h;
typedef __attribute__((ext_vector_type(8)))  float    v8f;
typedef __attribute__((ext_vector_type(4)))  unsigned u32x4;
typedef __attribute__((ext_vector_type(8)))  int      i32x8;
typedef __attribute__((ext_vector_type(4)))  int      i32x4;

__device__ __forceinline__ v8f wmma16x16x32(v16h a, v16h b, v8f c) {
  return __builtin_amdgcn_wmma_f32_16x16x32_f16(
      false, a, false, b, (short)0, c, false, false);
}

// A-fragment K index for element e, lane-half lh (ISA 16-bit A 16x32 layout).
__device__ __forceinline__ int a_kidx(int e, int lh) {
  return ((e >> 3) << 4) + (lh << 3) + (e & 7);
}

// Flat shared-aperture address truncated to 32 bits == LDS byte offset (ISA 10.2).
__device__ __forceinline__ unsigned lds_off(const void* p) {
  return (unsigned)(size_t)p;
}

// TDM: 1-D contiguous f16 tile (nelem elements) global -> LDS. Descriptor per
// CDNA5 ISA: count=1, type=2, data_size=2B,
// tensor_dim0 = tile_dim0 = nelem, tensor_dim1 = tile_dim1 = 1.
// clang-23 toolchain: 6-arg builtin (g0, g1, g2, g3, extra, cpol).
__device__ __forceinline__ void tdm_load_f16(unsigned ldsaddr, const _Float16* gsrc,
                                             unsigned nelem) {
  unsigned long long ga = (unsigned long long)(size_t)gsrc;
  u32x4 g0;
  g0[0] = 1u;                                   // count=1 (valid user D#)
  g0[1] = ldsaddr;                              // lds_addr
  g0[2] = (unsigned)(ga & 0xFFFFFFFFu);         // global_addr[31:0]
  g0[3] = (unsigned)((ga >> 32) & 0x1FFFFFFu)   // global_addr[56:32]
          | (2u << 30);                         // type=2 ("image")
  i32x8 g1;
  g1[0] = (int)(1u << 16);                      // workgroup_mask=0, data_size=1 (2B)
  g1[1] = (int)((nelem & 0xFFFFu) << 16);       // tensor_dim0[15:0]
  g1[2] = (int)((nelem >> 16) | (1u << 16));    // tensor_dim0[31:16], tensor_dim1=1
  g1[3] = (int)((nelem & 0xFFFFu) << 16);       // tile_dim0 = nelem
  g1[4] = 1;                                    // tile_dim1=1, tile_dim2=0
  g1[5] = (int)nelem;                           // tensor_dim0_stride[31:0]
  g1[6] = 0;
  g1[7] = 0;
  i32x4 z4 = {0, 0, 0, 0};
  i32x8 z8 = {0, 0, 0, 0, 0, 0, 0, 0};
  __builtin_amdgcn_tensor_load_to_lds(g0, g1, z4, z4, z8, 0);
}

// ---------------------------------------------------------------------------
// QKV projection: Y = X @ W^T + b (f32 in, f16 out in [B,H,S,DK] head layout).
// One wave per 16x16 tile, 4 waves/block across n-tiles.
// ---------------------------------------------------------------------------
__global__ void proj_qkv_kernel(const float* __restrict__ X,
                                const float* __restrict__ W,
                                const float* __restrict__ bias,
                                _Float16* __restrict__ Y) {
  const int lane = threadIdx.x & 31;
  const int wave = threadIdx.x >> 5;
  const int r    = lane & 15;
  const int lh   = lane >> 4;
  const int n0   = (blockIdx.x * 4 + wave) * 16;
  const int m0   = blockIdx.y * 16;

  v8f acc = {};
  for (int k0 = 0; k0 < DMODEL; k0 += 32) {
    v16h a, b;
    const float* xrow = X + (size_t)(m0 + r) * DMODEL + k0;
    const float* wrow = W + (size_t)(n0 + r) * DMODEL + k0 + lh * 16;
    if (k0 + 32 < DMODEL) {
      __builtin_prefetch(xrow + 32, 0, 1);
      __builtin_prefetch(wrow + 32, 0, 1);
    }
#pragma unroll
    for (int e = 0; e < 16; ++e) a[e] = (_Float16)xrow[lh * 16 + e];
#pragma unroll
    for (int e = 0; e < 16; ++e) b[e] = (_Float16)wrow[e];
    acc = wmma16x16x32(a, b, acc);
  }

  const int n  = n0 + r;
  const int h  = n >> 6;
  const int dk = n & 63;
  const float bv = bias[n];
#pragma unroll
  for (int g = 0; g < 8; ++g) {
    const int m  = m0 + g + (lh << 3);
    const int bb = m >> 11;
    const int s  = m & (SEQ - 1);
    Y[(((size_t)bb * NHEAD + h) * SEQ + s) * DHEAD + dk] = (_Float16)(acc[g] + bv);
  }
}

// ---------------------------------------------------------------------------
// Flash attention: 4 waves/block, each wave owns one 16-row query tile.
// K/V chunks (32 keys = 4KB each) staged to LDS by TDM, double-buffered.
// ---------------------------------------------------------------------------
__global__ void attn_kernel(const _Float16* __restrict__ Qh,
                            const _Float16* __restrict__ Kh,
                            const _Float16* __restrict__ Vh,
                            _Float16* __restrict__ Xout) {
  __shared__ __align__(16) _Float16 kbuf[2][KCHUNK * DHEAD];
  __shared__ __align__(16) _Float16 vbuf[2][KCHUNK * DHEAD];
  __shared__ __align__(16) _Float16 ldsP[4][16 * 32];

  const int lane = threadIdx.x & 31;
  const int wave = threadIdx.x >> 5;
  const int r    = lane & 15;
  const int lh   = lane >> 4;
  const int s0   = blockIdx.x * 64 + wave * 16;
  const int bh   = blockIdx.y;
  const int bb   = bh / NHEAD, head = bh % NHEAD;

  const _Float16* Q = Qh + (size_t)bh * SEQ * DHEAD;
  const _Float16* K = Kh + (size_t)bh * SEQ * DHEAD;
  const _Float16* V = Vh + (size_t)bh * SEQ * DHEAD;

  // Prologue: stage chunk 0.
  if (wave == 0) {
    tdm_load_f16(lds_off(&kbuf[0][0]), K, KCHUNK * DHEAD);
    tdm_load_f16(lds_off(&vbuf[0][0]), V, KCHUNK * DHEAD);
  }

  // Q fragments (two 16x32 A-fragments), 1/sqrt(64) folded in. Per-lane data is
  // two aligned 8-half runs -> 128-bit loads.
  v16h qa[2];
#pragma unroll
  for (int f = 0; f < 2; ++f) {
    const _Float16* qrow = Q + (size_t)(s0 + r) * DHEAD + f * 32;
    const v8h q0 = *(const v8h*)(qrow + lh * 16);
    const v8h q1 = *(const v8h*)(qrow + lh * 16 + 8);
#pragma unroll
    for (int e = 0; e < 8; ++e) {
      qa[f][e]     = (_Float16)((float)q0[e] * 0.125f);
      qa[f][e + 8] = (_Float16)((float)q1[e] * 0.125f);
    }
  }

  float mrow[8], lrow[8];
  v8f o[4] = {v8f{}, v8f{}, v8f{}, v8f{}};
#pragma unroll
  for (int g = 0; g < 8; ++g) { mrow[g] = -1e30f; lrow[g] = 0.0f; }

  for (int c = 0; c < NCHUNK; ++c) {
    const int kt = c * KCHUNK;
    if (wave == 0) {
      if (c < NCHUNK - 1) {
        const size_t nb = (size_t)(kt + KCHUNK) * DHEAD;
        tdm_load_f16(lds_off(&kbuf[(c + 1) & 1][0]), K + nb, KCHUNK * DHEAD);
        tdm_load_f16(lds_off(&vbuf[(c + 1) & 1][0]), V + nb, KCHUNK * DHEAD);
        __builtin_amdgcn_s_wait_tensorcnt(2);   // chunk c complete (in-order)
      } else {
        __builtin_amdgcn_s_wait_tensorcnt(0);
      }
    }
    __syncthreads();                            // chunk c visible to all waves
    const _Float16* kc = &kbuf[c & 1][0];
    const _Float16* vc = &vbuf[c & 1][0];

    // ---- scores: two 16x16 tiles over the 32-key chunk ----
    v8f sc0 = {}, sc1 = {};
#pragma unroll
    for (int f = 0; f < 2; ++f) {
      v16h bk0, bk1;
      const _Float16* k0row = kc + (size_t)r * DHEAD + f * 32 + lh * 16;
      const _Float16* k1row = kc + (size_t)(16 + r) * DHEAD + f * 32 + lh * 16;
      const v8h a0 = *(const v8h*)k0row, a1 = *(const v8h*)(k0row + 8);
      const v8h b0 = *(const v8h*)k1row, b1 = *(const v8h*)(k1row + 8);
#pragma unroll
      for (int e = 0; e < 8; ++e) {
        bk0[e] = a0[e]; bk0[e + 8] = a1[e];
        bk1[e] = b0[e]; bk1[e + 8] = b1[e];
      }
      sc0 = wmma16x16x32(qa[f], bk0, sc0);
      sc1 = wmma16x16x32(qa[f], bk1, sc1);
    }

    // ---- online softmax (rows g + 8*lh live in this 16-lane half) ----
    float p0[8], p1[8], scale_old[8];
#pragma unroll
    for (int g = 0; g < 8; ++g) {
      float cmax = fmaxf(sc0[g], sc1[g]);
#pragma unroll
      for (int msk = 1; msk <= 8; msk <<= 1)
        cmax = fmaxf(cmax, __shfl_xor(cmax, msk, 32));
      const float mn = fmaxf(mrow[g], cmax);
      scale_old[g] = __expf(mrow[g] - mn);
      mrow[g] = mn;
      p0[g] = __expf(sc0[g] - mn);
      p1[g] = __expf(sc1[g] - mn);
      float csum = p0[g] + p1[g];
#pragma unroll
      for (int msk = 1; msk <= 8; msk <<= 1)
        csum += __shfl_xor(csum, msk, 32);
      lrow[g] = lrow[g] * scale_old[g] + csum;
    }
#pragma unroll
    for (int t = 0; t < 4; ++t)
#pragma unroll
      for (int g = 0; g < 8; ++g) o[t][g] *= scale_old[g];

    // ---- P: C/D layout -> LDS [16][32] (per-wave region) ----
    _Float16* pw = &ldsP[wave][0];
#pragma unroll
    for (int g = 0; g < 8; ++g) {
      const int row = g + (lh << 3);
      pw[row * 32 + r]      = (_Float16)p0[g];
      pw[row * 32 + 16 + r] = (_Float16)p1[g];
    }
    __syncthreads();                            // P visible; kbuf reads done
    v16h pa;
    {
      const v8h pa0 = *(const v8h*)(pw + r * 32 + lh * 16);
      const v8h pa1 = *(const v8h*)(pw + r * 32 + lh * 16 + 8);
#pragma unroll
      for (int e = 0; e < 8; ++e) { pa[e] = pa0[e]; pa[e + 8] = pa1[e]; }
    }

    // ---- O += P(16x32) x V(32x64) ----
#pragma unroll
    for (int t = 0; t < 4; ++t) {
      v16h vb;
#pragma unroll
      for (int e = 0; e < 16; ++e)
        vb[e] = vc[(size_t)(lh * 16 + e) * DHEAD + t * 16 + r];
      o[t] = wmma16x16x32(pa, vb, o[t]);
    }
    __syncthreads();                            // buffer c&1 safe to overwrite
  }

  // ---- epilogue: O /= l, write [B,S,D] f16 ----
#pragma unroll
  for (int t = 0; t < 4; ++t) {
#pragma unroll
    for (int g = 0; g < 8; ++g) {
      const int row = g + (lh << 3);
      const int s   = s0 + row;
      const int dk  = t * 16 + r;
      Xout[((size_t)bb * SEQ + s) * DMODEL + head * DHEAD + dk] =
          (_Float16)(o[t][g] / lrow[g]);
    }
  }
}

// ---------------------------------------------------------------------------
// Output projection: out = X @ W_o^T + b_o (X f16 [M,768] -> f32 [M,768]).
// ---------------------------------------------------------------------------
__global__ void oproj_kernel(const _Float16* __restrict__ X,
                             const float* __restrict__ W,
                             const float* __restrict__ bias,
                             float* __restrict__ Y) {
  const int lane = threadIdx.x & 31;
  const int wave = threadIdx.x >> 5;
  const int r    = lane & 15;
  const int lh   = lane >> 4;
  const int n0   = (blockIdx.x * 4 + wave) * 16;
  const int m0   = blockIdx.y * 16;

  v8f acc = {};
  for (int k0 = 0; k0 < DMODEL; k0 += 32) {
    v16h a, b;
    const _Float16* xrow = X + (size_t)(m0 + r) * DMODEL + k0;
    const float*    wrow = W + (size_t)(n0 + r) * DMODEL + k0 + lh * 16;
    if (k0 + 32 < DMODEL) {
      __builtin_prefetch(xrow + 32, 0, 1);
      __builtin_prefetch(wrow + 32, 0, 1);
    }
    const v8h a0 = *(const v8h*)(xrow + lh * 16);
    const v8h a1 = *(const v8h*)(xrow + lh * 16 + 8);
#pragma unroll
    for (int e = 0; e < 8; ++e) { a[e] = a0[e]; a[e + 8] = a1[e]; }
#pragma unroll
    for (int e = 0; e < 16; ++e) b[e] = (_Float16)wrow[e];
    acc = wmma16x16x32(a, b, acc);
  }

  const int n = n0 + r;
  const float bv = bias[n];
#pragma unroll
  for (int g = 0; g < 8; ++g) {
    const int m = m0 + g + (lh << 3);
    Y[(size_t)m * DMODEL + n] = acc[g] + bv;
  }
}

// ---------------------------------------------------------------------------
extern "C" void kernel_launch(void* const* d_in, const int* in_sizes, int n_in,
                              void* d_out, int out_size, void* d_ws, size_t ws_size,
                              hipStream_t stream) {
  (void)in_sizes; (void)n_in; (void)out_size; (void)ws_size;
  const float* q   = (const float*)d_in[0];
  const float* k   = (const float*)d_in[1];
  const float* v   = (const float*)d_in[2];
  const float* w_q = (const float*)d_in[3];
  const float* b_q = (const float*)d_in[4];
  const float* w_k = (const float*)d_in[5];
  const float* b_k = (const float*)d_in[6];
  const float* w_v = (const float*)d_in[7];
  const float* b_v = (const float*)d_in[8];
  const float* w_o = (const float*)d_in[9];
  const float* b_o = (const float*)d_in[10];

  const size_t headElems = (size_t)BATCH * NHEAD * SEQ * DHEAD;
  _Float16* qh   = (_Float16*)d_ws;
  _Float16* kh   = qh + headElems;
  _Float16* vh   = kh + headElems;
  _Float16* xatt = vh + headElems;

  const dim3 gemmGrid(12, MROWS / 16);
  const dim3 gemmBlock(128);

  proj_qkv_kernel<<<gemmGrid, gemmBlock, 0, stream>>>(q, w_q, b_q, qh);
  proj_qkv_kernel<<<gemmGrid, gemmBlock, 0, stream>>>(k, w_k, b_k, kh);
  proj_qkv_kernel<<<gemmGrid, gemmBlock, 0, stream>>>(v, w_v, b_v, vh);

  attn_kernel<<<dim3(SEQ / 64, BATCH * NHEAD), 128, 0, stream>>>(qh, kh, vh, xatt);

  oproj_kernel<<<gemmGrid, gemmBlock, 0, stream>>>(xatt, w_o, b_o, (float*)d_out);
}
